// AttentionTagClassifier_21844203668211
// MI455X (gfx1250) — matrix operand order repaired and checked by reading, not compiled, not run on hardware
//
#include <hip/hip_runtime.h>
#include <hip/hip_bf16.h>

typedef __attribute__((ext_vector_type(16))) __bf16 v16bf;
typedef __attribute__((ext_vector_type(8)))  float  v8f;
typedef __attribute__((ext_vector_type(8)))  unsigned int v8u;
typedef __attribute__((ext_vector_type(4)))  unsigned int v4u;
typedef __attribute__((ext_vector_type(4)))  float  v4f;

// ---------------------------------------------------------------------------
// helpers
// ---------------------------------------------------------------------------
__device__ __forceinline__ float sigm(float x) {
    return 1.0f / (1.0f + __expf(-x));
}

// A fragment: 16x32 bf16 slice of A (M x K bf16, row-major, lda elements).
// ISA 7.12.2 layout: lane L (g=L/16, m=L%16):
//   V0..3 : K = g*8 + {0..7};  V4..7 : K = 16 + g*8 + {0..7}
// -> two contiguous 16-byte runs per lane.
__device__ __forceinline__ v16bf load_afrag(const __bf16* A, int lda, int m0, int k0) {
    const int lane = threadIdx.x & 31;
    const int g = lane >> 4;
    const __bf16* row = A + (size_t)(m0 + (lane & 15)) * lda + k0 + g * 8;
    v4u lo = *(const v4u*)(row);
    v4u hi = *(const v4u*)(row + 16);
    v8u u = { lo[0], lo[1], lo[2], lo[3], hi[0], hi[1], hi[2], hi[3] };
    return __builtin_bit_cast(v16bf, u);
}

// B fragment: 32x16 bf16 (K x N) with B[k][n] = W[n][k], W (N x K) row-major.
// Lane L (g=L/16, n=L%16) holds K = g*8 + {0..7} and 16 + g*8 + {0..7}: two contiguous 16-byte runs.
__device__ __forceinline__ v16bf load_bfrag(const __bf16* W, int ldw, int n0, int k0) {
    const int lane = threadIdx.x & 31;
    const int g = lane >> 4;
    const __bf16* row = W + (size_t)(n0 + (lane & 15)) * ldw + k0 + g * 8;
    v4u lo = *(const v4u*)(row);
    v4u hi = *(const v4u*)(row + 16);
    v8u u = { lo[0], lo[1], lo[2], lo[3], hi[0], hi[1], hi[2], hi[3] };
    return __builtin_bit_cast(v16bf, u);
}

#define WMMA_BF16(a, b, c) \
    __builtin_amdgcn_wmma_f32_16x16x32_bf16(false, (a), false, (b), (short)0, (c), false, false)

// ---------------------------------------------------------------------------
// WMMA GEMM:  C[M,N](f32) = A[M,K](bf16) @ W[N,K](bf16)^T (+bias[N]) (+Cadd)
// One wave per 16x64 tile group: A-fragment reused across 4 WMMAs.
// ---------------------------------------------------------------------------
__global__ void wmma_gemm(const __bf16* __restrict__ A, int lda,
                          const __bf16* __restrict__ W, int ldw,
                          const float* __restrict__ bias,
                          const float* __restrict__ Cadd, int ldadd,
                          float* __restrict__ C, int ldc,
                          int M, int N, int K) {
    const int lane = threadIdx.x & 31;
    const int wave = threadIdx.x >> 5;
    const int wpb  = blockDim.x >> 5;
    const int grpN = N >> 6;                  // groups of 4 N-tiles
    const int nGrp = (M >> 4) * grpN;

    for (int grp = blockIdx.x * wpb + wave; grp < nGrp; grp += gridDim.x * wpb) {
        const int m0 = (grp / grpN) << 4;
        const int n0 = (grp % grpN) << 6;
        v8f acc0 = {}, acc1 = {}, acc2 = {}, acc3 = {};
        for (int k0 = 0; k0 < K; k0 += 32) {
            if (k0 + 32 < K)   // gfx1250 global_prefetch_b8 on next B K-slab
                __builtin_prefetch(W + (size_t)(n0 + (lane & 15)) * ldw + k0 + 32, 0, 1);
            v16bf a = load_afrag(A, lda, m0, k0);
            acc0 = WMMA_BF16(a, load_bfrag(W, ldw, n0,      k0), acc0);
            acc1 = WMMA_BF16(a, load_bfrag(W, ldw, n0 + 16, k0), acc1);
            acc2 = WMMA_BF16(a, load_bfrag(W, ldw, n0 + 32, k0), acc2);
            acc3 = WMMA_BF16(a, load_bfrag(W, ldw, n0 + 48, k0), acc3);
        }
        const int g = lane >> 4;
        const int cb = lane & 15;
        v8f accs[4] = { acc0, acc1, acc2, acc3 };
#pragma unroll
        for (int q = 0; q < 4; ++q) {
            const int col = n0 + q * 16 + cb;
            const float bv = bias ? bias[col] : 0.0f;
#pragma unroll
            for (int r = 0; r < 8; ++r) {
                const int row = m0 + g * 8 + r;     // C layout: VGPR r -> M = g*8+r
                float v = accs[q][r] + bv;
                if (Cadd) v += Cadd[(size_t)row * ldadd + col];
                C[(size_t)row * ldc + col] = v;
            }
        }
    }
}

// ---------------------------------------------------------------------------
// conversion / pointwise kernels
// ---------------------------------------------------------------------------
__global__ void cvt_bf16(const float* __restrict__ src, __bf16* __restrict__ dst, int n4) {
    // n4 = n/4; every buffer size here is a multiple of 4
    for (int i = blockIdx.x * blockDim.x + threadIdx.x; i < n4; i += gridDim.x * blockDim.x) {
        v4f x = *(const v4f*)(src + (size_t)i * 4);
        __bf16* d = dst + (size_t)i * 4;
        d[0] = (__bf16)x[0]; d[1] = (__bf16)x[1];
        d[2] = (__bf16)x[2]; d[3] = (__bf16)x[3];
    }
}

// attn_W (1024x1024 f32) -> transposed bf16 copy
__global__ void transpose1024_bf(const float* __restrict__ src, __bf16* __restrict__ dst) {
    __shared__ float tile[16][17];
    int tx = threadIdx.x & 15, ty = threadIdx.x >> 4;
    int x = blockIdx.x * 16 + tx;
    int y = blockIdx.y * 16 + ty;
    tile[ty][tx] = src[(size_t)y * 1024 + x];
    __syncthreads();
    int ox = blockIdx.y * 16 + tx;
    int oy = blockIdx.x * 16 + ty;
    dst[(size_t)oy * 1024 + ox] = (__bf16)tile[tx][ty];
}

// encoder LSTM pointwise, both directions via blockIdx.y. H=512, T=64, B=32.
__global__ void enc_lstm_elem(const float* __restrict__ gates_f,
                              const float* __restrict__ gates_b,
                              float* __restrict__ c_f, float* __restrict__ c_b,
                              __bf16* __restrict__ h_f_bf, __bf16* __restrict__ h_b_bf,
                              const int* __restrict__ mask,
                              float* __restrict__ output, int t) {
    int i = blockIdx.x * blockDim.x + threadIdx.x;
    if (i >= 32 * 512) return;
    const int dir = blockIdx.y;
    const float* g = dir ? gates_b : gates_f;
    float* c = dir ? c_b : c_f;
    __bf16* hbf = dir ? h_b_bf : h_f_bf;
    const int tt = dir ? (63 - t) : t;
    const int b = i >> 9, j = i & 511;
    const float* gr = g + (size_t)b * 2048;
    float ii = gr[j], ff = gr[512 + j], gg = gr[1024 + j], oo = gr[1536 + j];
    float cn = sigm(ff) * c[i] + sigm(ii) * tanhf(gg);
    float hn = sigm(oo) * tanhf(cn);
    c[i] = cn;
    hbf[i] = (__bf16)hn;                   // bf16 h feeds next step's WMMA GEMM
    float m = (float)mask[b * 64 + tt];
    output[((size_t)b * 64 + tt) * 1024 + dir * 512 + j] = hn * m;
}

__global__ void gather_last(const float* __restrict__ output,
                            const int* __restrict__ mask,
                            __bf16* __restrict__ last_out_bf) {
    int b = blockIdx.x;
    __shared__ int len;
    if (threadIdx.x == 0) {
        int s = 0;
        for (int t = 0; t < 64; ++t) s += mask[b * 64 + t];
        len = s - 1;
    }
    __syncthreads();
    for (int e = threadIdx.x; e < 1024; e += blockDim.x)
        last_out_bf[(size_t)b * 1024 + e] =
            (__bf16)output[((size_t)b * 64 + len) * 1024 + e];
}

// sc = hW . output[b,t,:] + attn_b ; softmax over t ; *mask ; context = sum a_t*output
__global__ void attend_kernel(const float* __restrict__ hW,
                              const float* __restrict__ output,
                              const int* __restrict__ mask,
                              const float* __restrict__ attn_b,
                              float* __restrict__ context) {
    __shared__ float sc[64];
    const int b = blockIdx.x, tid = threadIdx.x;
    if (tid < 64) {
        const float* o = output + ((size_t)b * 64 + tid) * 1024;
        const float* w = hW + (size_t)b * 1024;
        float s = 0.0f;
        for (int e = 0; e < 1024; ++e) s += w[e] * o[e];
        sc[tid] = s + attn_b[0];
    }
    __syncthreads();
    if (tid == 0) {
        float mx = sc[0];
        for (int t = 1; t < 64; ++t) mx = fmaxf(mx, sc[t]);
        float sum = 0.0f;
        for (int t = 0; t < 64; ++t) { float e = __expf(sc[t] - mx); sc[t] = e; sum += e; }
        float inv = 1.0f / sum;
        for (int t = 0; t < 64; ++t) sc[t] = sc[t] * inv * (float)mask[b * 64 + t];
    }
    __syncthreads();
    for (int e = tid; e < 1024; e += blockDim.x) {
        float acc = 0.0f;
        const float* o = output + (size_t)b * 64 * 1024 + e;
        for (int t = 0; t < 64; ++t) acc += sc[t] * o[(size_t)t * 1024];
        context[(size_t)b * 1024 + e] = acc;
    }
}

// dec_input(bf16) = [emb(512) | context(1024) | output[:,t,:](1024)]
__global__ void dec_assemble(const __bf16* __restrict__ tag_embed_bf,
                             const int* __restrict__ last_tag,
                             const float* __restrict__ context,
                             const float* __restrict__ output, int t,
                             __bf16* __restrict__ dec_input) {
    int b = blockIdx.x;
    for (int j = threadIdx.x; j < 2560; j += blockDim.x) {
        __bf16 v;
        if (j < 512)
            v = (t == 0) ? (__bf16)0.0f : tag_embed_bf[(size_t)last_tag[b] * 512 + j];
        else if (j < 1536)
            v = (__bf16)context[(size_t)b * 1024 + (j - 512)];
        else
            v = (__bf16)output[((size_t)b * 64 + t) * 1024 + (j - 1536)];
        dec_input[(size_t)b * 2560 + j] = v;
    }
}

// decoder LSTM pointwise (Hd=1024) + build concated(bf16)=[h, context_old]
__global__ void dec_lstm_elem(const float* __restrict__ gates,
                              float* __restrict__ c,
                              __bf16* __restrict__ h_bf,
                              const float* __restrict__ context,
                              __bf16* __restrict__ concated) {
    int i = blockIdx.x * blockDim.x + threadIdx.x;
    if (i >= 32 * 1024) return;
    const int b = i >> 10, j = i & 1023;
    const float* g = gates + (size_t)b * 4096;
    float ii = g[j], ff = g[1024 + j], gg = g[2048 + j], oo = g[3072 + j];
    float cn = sigm(ff) * c[i] + sigm(ii) * tanhf(gg);
    float hn = sigm(oo) * tanhf(cn);
    c[i] = cn;
    h_bf[i] = (__bf16)hn;
    concated[(size_t)b * 2048 + j] = (__bf16)hn;
    concated[(size_t)b * 2048 + 1024 + j] = (__bf16)context[i];
}

__global__ void argmax_kernel(const float* __restrict__ score, int t,
                              int* __restrict__ last_tag) {
    const int b = blockIdx.x;
    const float* s = score + ((size_t)b * 64 + t) * 2048;
    __shared__ float bv[256];
    __shared__ int   bi[256];
    float best = -3.402823e38f; int bestI = 0;
    for (int v = threadIdx.x; v < 2048; v += blockDim.x) {
        float x = s[v];
        if (x > best) { best = x; bestI = v; }
    }
    bv[threadIdx.x] = best; bi[threadIdx.x] = bestI;
    __syncthreads();
    for (int st = 128; st > 0; st >>= 1) {
        if (threadIdx.x < st) {
            if (bv[threadIdx.x + st] > bv[threadIdx.x] ||
                (bv[threadIdx.x + st] == bv[threadIdx.x] && bi[threadIdx.x + st] < bi[threadIdx.x])) {
                bv[threadIdx.x] = bv[threadIdx.x + st];
                bi[threadIdx.x] = bi[threadIdx.x + st];
            }
        }
        __syncthreads();
    }
    if (threadIdx.x == 0) last_tag[b] = bi[0];
}

// ---------------------------------------------------------------------------
// host side
// ---------------------------------------------------------------------------
static inline void gemm(hipStream_t s,
                        const __bf16* A, int lda, const __bf16* W, int ldw,
                        const float* bias, const float* Cadd, int ldadd,
                        float* C, int ldc, int M, int N, int K) {
    int groups = (M / 16) * (N / 64);
    int blocks = (groups + 3) / 4;          // 4 waves (128 threads) per block
    if (blocks > 4096) blocks = 4096;
    wmma_gemm<<<blocks, 128, 0, s>>>(A, lda, W, ldw, bias, Cadd, ldadd, C, ldc, M, N, K);
}

static inline void cvt(hipStream_t s, const float* src, __bf16* dst, size_t n) {
    int n4 = (int)(n / 4);
    int blocks = (n4 + 255) / 256;
    if (blocks > 2048) blocks = 2048;
    cvt_bf16<<<blocks, 256, 0, s>>>(src, dst, n4);
}

extern "C" void kernel_launch(void* const* d_in, const int* in_sizes, int n_in,
                              void* d_out, int out_size, void* d_ws, size_t ws_size,
                              hipStream_t stream) {
    constexpr int B = 32, T = 64, D = 512, H = 512, G = 2048;
    constexpr int Hd = 1024, G2 = 4096, V = 2048, DIN = 2560;

    const float* emb       = (const float*)d_in[0];
    const int*   mask      = (const int*)  d_in[1];
    const float* enc_Wih_f = (const float*)d_in[2];
    const float* enc_Whh_f = (const float*)d_in[3];
    const float* enc_bih_f = (const float*)d_in[4];
    const float* enc_bhh_f = (const float*)d_in[5];
    const float* enc_Wih_b = (const float*)d_in[6];
    const float* enc_Whh_b = (const float*)d_in[7];
    const float* enc_bih_b = (const float*)d_in[8];
    const float* enc_bhh_b = (const float*)d_in[9];
    const float* attn_W    = (const float*)d_in[10];
    const float* attn_b    = (const float*)d_in[11];
    const float* tag_embed = (const float*)d_in[12];
    const float* dec_Wih   = (const float*)d_in[13];
    const float* dec_Whh   = (const float*)d_in[14];
    const float* dec_bih   = (const float*)d_in[15];
    const float* dec_bhh   = (const float*)d_in[16];
    const float* Wout      = (const float*)d_in[17];
    const float* bout      = (const float*)d_in[18];
    float* out = (float*)d_out;

    // ---- workspace carve-up; everything padded to 32-byte boundaries ----
    char* base = (char*)d_ws;
    size_t off = 0;
    auto carveF = [&](size_t n) {                       // n floats
        float* p = (float*)(base + off);
        off += ((n * 4 + 31) & ~(size_t)31);
        return p;
    };
    auto carveB = [&](size_t n) {                       // n bf16
        __bf16* p = (__bf16*)(base + off);
        off += ((n * 2 + 31) & ~(size_t)31);
        return p;
    };

    // f32 scratch
    float* xg_f     = carveF((size_t)B * T * G);
    float* xg_b     = carveF((size_t)B * T * G);
    float* output   = carveF((size_t)B * T * Hd);
    float* c_states = carveF(2 * (size_t)B * H + (size_t)B * Hd);  // c_f, c_b, c_d
    float* c_f = c_states;
    float* c_b = c_f + (size_t)B * H;
    float* c_d = c_b + (size_t)B * H;
    float* gates_f   = carveF((size_t)B * G);
    float* gates_b   = carveF((size_t)B * G);
    float* hW        = carveF((size_t)B * Hd);
    float* context   = carveF((size_t)B * Hd);
    float* dec_gates = carveF((size_t)B * G2);
    int*   last_tag  = (int*)carveF(64);

    // bf16 weight copies (converted once per launch, stay hot in 192MB L2)
    __bf16* emb_bf       = carveB((size_t)B * T * D);
    __bf16* enc_Wih_f_bf = carveB((size_t)G * D);
    __bf16* enc_Wih_b_bf = carveB((size_t)G * D);
    __bf16* enc_Whh_f_bf = carveB((size_t)G * H);
    __bf16* enc_Whh_b_bf = carveB((size_t)G * H);
    __bf16* attn_WT_bf   = carveB((size_t)Hd * Hd);
    __bf16* tag_embed_bf = carveB((size_t)V * 512);
    __bf16* dec_Wih_bf   = carveB((size_t)G2 * DIN);
    __bf16* dec_Whh_bf   = carveB((size_t)G2 * Hd);
    __bf16* Wout_bf      = carveB((size_t)V * 2048);

    // bf16 activations (zero-init h states each call)
    __bf16* h_bf_states = carveB(2 * (size_t)B * H + (size_t)B * Hd); // h_f, h_b, h_d
    __bf16* h_f_bf = h_bf_states;
    __bf16* h_b_bf = h_f_bf + (size_t)B * H;
    __bf16* h_d_bf = h_b_bf + (size_t)B * H;
    __bf16* last_out_bf  = carveB((size_t)B * Hd);
    __bf16* dec_input_bf = carveB((size_t)B * DIN);
    __bf16* concated_bf  = carveB((size_t)B * 2048);

    // deterministic per-call init of recurrent state
    hipMemsetAsync(c_states, 0, (2 * (size_t)B * H + (size_t)B * Hd) * sizeof(float), stream);
    hipMemsetAsync(h_bf_states, 0, (2 * (size_t)B * H + (size_t)B * Hd) * sizeof(__bf16), stream);
    hipMemsetAsync(last_tag, 0, B * sizeof(int), stream);

    // ---- one-time conversions ----
    cvt(stream, emb,       emb_bf,       (size_t)B * T * D);
    cvt(stream, enc_Wih_f, enc_Wih_f_bf, (size_t)G * D);
    cvt(stream, enc_Wih_b, enc_Wih_b_bf, (size_t)G * D);
    cvt(stream, enc_Whh_f, enc_Whh_f_bf, (size_t)G * H);
    cvt(stream, enc_Whh_b, enc_Whh_b_bf, (size_t)G * H);
    cvt(stream, tag_embed, tag_embed_bf, (size_t)V * 512);
    cvt(stream, dec_Wih,   dec_Wih_bf,   (size_t)G2 * DIN);
    cvt(stream, dec_Whh,   dec_Whh_bf,   (size_t)G2 * Hd);
    cvt(stream, Wout,      Wout_bf,      (size_t)V * 2048);
    transpose1024_bf<<<dim3(64, 64), 256, 0, stream>>>(attn_W, attn_WT_bf);

    // ---- Phase A: xg = emb @ Wih^T + bih (2048x512 -> 2048), both dirs ----
    gemm(stream, emb_bf, D, enc_Wih_f_bf, D, enc_bih_f, nullptr, 0, xg_f, G, B * T, G, D);
    gemm(stream, emb_bf, D, enc_Wih_b_bf, D, enc_bih_b, nullptr, 0, xg_b, G, B * T, G, D);

    // ---- Encoder scan (fwd at t, bwd at 63-t, independent within a step) ----
    for (int t = 0; t < T; ++t) {
        gemm(stream, h_f_bf, H, enc_Whh_f_bf, H, enc_bhh_f,
             xg_f + (size_t)t * G, T * G, gates_f, G, B, G, H);
        gemm(stream, h_b_bf, H, enc_Whh_b_bf, H, enc_bhh_b,
             xg_b + (size_t)(T - 1 - t) * G, T * G, gates_b, G, B, G, H);
        enc_lstm_elem<<<dim3((B * H + 255) / 256, 2), 256, 0, stream>>>(
            gates_f, gates_b, c_f, c_b, h_f_bf, h_b_bf, mask, output, t);
    }

    // ---- context0 from last_output ----
    gather_last<<<B, 256, 0, stream>>>(output, mask, last_out_bf);
    gemm(stream, last_out_bf, Hd, attn_WT_bf, Hd, nullptr, nullptr, 0, hW, Hd, B, Hd, Hd);
    attend_kernel<<<B, 256, 0, stream>>>(hW, output, mask, attn_b, context);

    // ---- Decoder scan ----
    for (int t = 0; t < T; ++t) {
        dec_assemble<<<B, 256, 0, stream>>>(tag_embed_bf, last_tag, context, output, t,
                                            dec_input_bf);
        gemm(stream, dec_input_bf, DIN, dec_Wih_bf, DIN, dec_bih, nullptr, 0,
             dec_gates, G2, B, G2, DIN);
        gemm(stream, h_d_bf, Hd, dec_Whh_bf, Hd, dec_bhh, dec_gates, G2,
             dec_gates, G2, B, G2, Hd);
        dec_lstm_elem<<<(B * Hd + 255) / 256, 256, 0, stream>>>(
            dec_gates, c_d, h_d_bf, context, concated_bf);
        // score -> d_out[b, t, :]  (row stride T*V, base offset t*V)
        gemm(stream, concated_bf, 2048, Wout_bf, 2048, bout, nullptr, 0,
             out + (size_t)t * V, T * V, B, V, 2048);
        // new context from h
        gemm(stream, h_d_bf, Hd, attn_WT_bf, Hd, nullptr, nullptr, 0, hW, Hd, B, Hd, Hd);
        attend_kernel<<<B, 256, 0, stream>>>(hW, output, mask, attn_b, context);
        argmax_kernel<<<B, 256, 0, stream>>>(out, t, last_tag);
    }
}
